// WindowAttention_85495618994226
// MI455X (gfx1250) — compile-verified, matrix operand order checked by reading, not run on hardware
//
#include <hip/hip_runtime.h>
#include <hip/hip_bf16.h>

typedef __attribute__((ext_vector_type(16))) _Float16 v16h;
typedef __attribute__((ext_vector_type(8)))  _Float16 v8h;
typedef __attribute__((ext_vector_type(8)))  float    v8f;
typedef __attribute__((ext_vector_type(4)))  float    v4f;

#define NWIN   1024
#define NTOK   64
#define CDIM   192
#define NHEAD  6
#define HDIM   32
#define QKSCALE 0.17677669529663687f   // 32^-0.5
#define EPSV   1.1920929e-07f

// ---- LDS layout (single block => stable 32-bit LDS offsets for async ops) ----
#define OFF_NX     0u                       // norm_x f16          24576
#define OFF_O16    24576u                   // attn out f16        24576
#define OFF_MASK   49152u                   // window mask f16      8192
#define OFF_WSLCA  57344u                   // weight slice buf A  36864
#define OFF_WSLCB  94208u                   // weight slice buf B  36864
#define OFF_WP     57344u                   // Wp f16 (proj phase, aliases A+B) 73728
#define OFF_QH     131072u                  // 64x32 f16            4096
#define OFF_KH     135168u
#define OFF_VHT    139264u                  // 32x64 f16 (d-major)
#define OFF_S32    143360u                  // 64x64 f32           16384 (ps16 aliases)
#define OFF_RED    159744u                  // 64x4 f32             1024
#define OFF_RSTAT  160768u                  // rinv[64]              256
#define OFF_RNORM  161024u                  // norm[64]              256
#define OFF_MWK    161280u                  // key validity[64]      256
#define OFF_MWSUM  161536u
#define SMEM_BYTES 161540u

// D = A(16x32 f16) * B(32x16 f16) + C(16x16 f32)
__device__ __forceinline__ v8f wmma16(v16h a, v16h b, v8f c) {
  return __builtin_amdgcn_wmma_f32_16x16x32_f16(false, a, false, b, (short)0, c, false, false);
}

// Fragment load (A and B share the layout): row = row0 + (lane&15);
// K elements = two contiguous 8-half chunks at k0 + hi*8 and k0 + hi*8 + 16.
__device__ __forceinline__ v16h ldfrag(const _Float16* p, int row0, int k0,
                                       int stride, int lo, int hi) {
  const _Float16* rp = p + (row0 + lo) * stride + k0 + hi * 8;
  v8h c0 = *(const v8h*)(rp);
  v8h c1 = *(const v8h*)(rp + 16);
  return __builtin_shufflevector(c0, c1, 0,1,2,3,4,5,6,7,8,9,10,11,12,13,14,15);
}

// CDNA5 async global->LDS copy, 16B per lane, tracked by ASYNCcnt.
__device__ __forceinline__ void async_b128(unsigned lds_addr, unsigned goff,
                                           const void* gbase) {
  asm volatile("global_load_async_to_lds_b128 %0, %1, %2"
               :: "v"(lds_addr), "v"(goff), "s"(gbase) : "memory");
}
__device__ __forceinline__ void wait_async_le9() {
  asm volatile("s_wait_asynccnt 0x9" ::: "memory");
}
__device__ __forceinline__ void wait_async_0() {
  asm volatile("s_wait_asynccnt 0x0" ::: "memory");
}

// Issue one head's Wq/Wk/Wv f16 slices (3 x 12288 B) : 9 async ops per thread.
__device__ __forceinline__ void issue_slice(const _Float16* ws16, int h,
                                            unsigned dstoff, int tid) {
  #pragma unroll
  for (int m = 0; m < 3; ++m) {
    const void* src = ws16 + m * (CDIM * CDIM) + h * (HDIM * CDIM);
    unsigned ldsm = dstoff + (unsigned)m * (HDIM * CDIM * 2);
    #pragma unroll
    for (int j = 0; j < 3; ++j) {
      unsigned c = (unsigned)(tid + 256 * j) * 16u;
      async_b128(ldsm + c, c, src);
    }
  }
}

// Pre-kernel: convert the four 192x192 weight matrices to f16 once.
__global__ __launch_bounds__(256) void cvt_weights(
    const float* __restrict__ Wq, const float* __restrict__ Wk,
    const float* __restrict__ Wv, const float* __restrict__ Wp,
    _Float16* __restrict__ ws16)
{
  const int i = blockIdx.x * 256 + threadIdx.x;   // over 4*9216 float4s
  const float* srcs[4] = {Wq, Wk, Wv, Wp};
  const int m = i / 9216, j = i % 9216;
  v4f f = ((const v4f*)srcs[m])[j];
  _Float16* d = ws16 + m * (CDIM * CDIM) + 4 * j;
  d[0] = (_Float16)f[0]; d[1] = (_Float16)f[1];
  d[2] = (_Float16)f[2]; d[3] = (_Float16)f[3];
}

__global__ __launch_bounds__(256) void win_attn_kernel(
    const float* __restrict__ x,   const float* __restrict__ mask,
    const float* __restrict__ mw,
    const float* __restrict__ bq,  const float* __restrict__ bk,
    const float* __restrict__ bv,  const float* __restrict__ bp,
    const _Float16* __restrict__ ws16,
    float* __restrict__ out, float* __restrict__ mwout)
{
  __shared__ __align__(16) char smem[SMEM_BYTES];

  _Float16* nxs16  = (_Float16*)(smem + OFF_NX);
  _Float16* o16    = (_Float16*)(smem + OFF_O16);
  _Float16* mask16 = (_Float16*)(smem + OFF_MASK);
  _Float16* qh     = (_Float16*)(smem + OFF_QH);
  _Float16* kh     = (_Float16*)(smem + OFF_KH);
  _Float16* vhT    = (_Float16*)(smem + OFF_VHT);
  float*    s32    = (float*)   (smem + OFF_S32);
  _Float16* ps16   = (_Float16*)(smem + OFF_S32);   // aliases s32 (safe: see softmax)
  _Float16* wp_s   = (_Float16*)(smem + OFF_WP);
  float*    red    = (float*)   (smem + OFF_RED);
  float*    rstat  = (float*)   (smem + OFF_RSTAT);
  float*    rnorm  = (float*)   (smem + OFF_RNORM);
  float*    mwk    = (float*)   (smem + OFF_MWK);
  float*    mwsum  = (float*)   (smem + OFF_MWSUM);

  const int b    = blockIdx.x;
  const int tid  = threadIdx.x;
  const int lane = tid & 31;
  const int wv   = tid >> 5;
  const int lo   = lane & 15;
  const int hi   = lane >> 4;

  // Kick off head-0 weight-slice prefetch immediately (overlaps Phase A).
  issue_slice(ws16, 0, OFF_WSLCA, tid);

  // ---------------- Phase A: x load, row norms, mask preload ----------------
  {
    const float* xg   = x + (size_t)b * NTOK * CDIM;
    const int row  = tid >> 2;          // 4 threads per token row
    const int part = tid & 3;           // 48 channels each
    float xr[48];
    float ss = 0.f;
    const float* xrow = xg + row * CDIM + part * 48;
    #pragma unroll
    for (int i = 0; i < 48; ++i) { float v = xrow[i]; xr[i] = v; ss += v * v; }
    red[row * 4 + part] = ss;
    if (tid < NTOK) mwk[tid] = mw[(size_t)b * NTOK + tid];
    // window mask -> f16 LDS (values 0 / -100, exact in f16)
    {
      const v4f* mg = (const v4f*)(mask + (size_t)(b & (NWIN - 1)) * NTOK * NTOK);
      for (int i = tid; i < 1024; i += 256) {
        v4f f = mg[i];
        mask16[4*i+0] = (_Float16)f[0]; mask16[4*i+1] = (_Float16)f[1];
        mask16[4*i+2] = (_Float16)f[2]; mask16[4*i+3] = (_Float16)f[3];
      }
    }
    __syncthreads();
    if (tid < NTOK) {
      float s = red[tid*4] + red[tid*4+1] + red[tid*4+2] + red[tid*4+3];
      float nv = fmaxf(__builtin_sqrtf(s), EPSV);
      rnorm[tid] = nv;
      rstat[tid] = 1.f / nv;
    }
    __syncthreads();
    const float rinv = rstat[row];
    _Float16* nxd = nxs16 + row * CDIM + part * 48;
    #pragma unroll
    for (int i = 0; i < 48; ++i) nxd[i] = (_Float16)(xr[i] * rinv);
    if (tid == 0) {
      float s = 0.f;
      for (int i = 0; i < NTOK; ++i) s += mwk[i];
      *mwsum = fminf(fmaxf(s, 0.f), 1.f);
    }
    __syncthreads();
  }

  const float* bsrc[3] = {bq, bk, bv};

  // ---------------- Per-head attention (double-buffered weight slices) -------
  for (int h = 0; h < NHEAD; ++h) {
    const unsigned curoff = (h & 1) ? OFF_WSLCB : OFF_WSLCA;
    if (h < NHEAD - 1) {
      issue_slice(ws16, h + 1, (h & 1) ? OFF_WSLCA : OFF_WSLCB, tid);
      wait_async_le9();       // current batch done (in-order), next still in flight
    } else {
      wait_async_0();
    }
    __syncthreads();
    const _Float16* wslc = (const _Float16*)(smem + curoff);

    // Q/K/V projections: 24 tile-jobs (3 mats x 4 Mtiles x 2 Ntiles), 6 k-steps
    for (int job = wv; job < 24; job += 8) {
      const int mat = job >> 3;
      const int mt  = (job & 7) >> 1;
      const int nt  = job & 1;
      const _Float16* Bs = wslc + mat * (HDIM * CDIM);
      v8f acc = {};
      #pragma unroll
      for (int ks = 0; ks < 6; ++ks) {
        v16h a  = ldfrag(nxs16, mt * 16, ks * 32, CDIM, lo, hi);
        v16h bf = ldfrag(Bs,    nt * 16, ks * 32, CDIM, lo, hi);
        acc = wmma16(a, bf, acc);
      }
      const float bias = bsrc[mat][h * HDIM + nt * 16 + lo];
      if (mat == 2) {   // V = ||x|| * (nx @ Wv) + bv, stored transposed vhT[d][tok]
        #pragma unroll
        for (int r = 0; r < 8; ++r) {
          const int tok = mt * 16 + r + hi * 8;
          vhT[(nt*16 + lo) * NTOK + tok] = (_Float16)(acc[r] * rnorm[tok] + bias);
        }
      } else {
        _Float16* dst = (mat == 0) ? qh : kh;
        #pragma unroll
        for (int r = 0; r < 8; ++r)
          dst[(mt*16 + r + hi*8) * HDIM + nt*16 + lo] = (_Float16)(acc[r] + bias);
      }
    }
    __syncthreads();

    // S = Q Kt : 16 tiles, single K=32 step each
    for (int job = wv; job < 16; job += 8) {
      const int mt = job >> 2, nt = job & 3;
      v16h a  = ldfrag(qh, mt * 16, 0, HDIM, lo, hi);
      v16h bf = ldfrag(kh, nt * 16, 0, HDIM, lo, hi);
      v8f acc = {};
      acc = wmma16(a, bf, acc);
      #pragma unroll
      for (int r = 0; r < 8; ++r)
        s32[(mt*16 + r + hi*8) * NTOK + nt*16 + lo] = acc[r];
    }
    __syncthreads();

    // Softmax with fused masks: 4 threads per row, 16 keys each (f32).
    // All s32 reads happen before the first barrier, so ps16 may alias s32.
    {
      const int row = tid & 63, q4 = tid >> 6;
      const float* srow = s32 + row * NTOK;
      const _Float16* mrow = mask16 + row * NTOK;
      float sv[16];
      float mx = -1e30f;
      #pragma unroll
      for (int i = 0; i < 16; ++i) {
        const int m = q4 * 16 + i;
        float v = srow[m] * QKSCALE + (float)mrow[m] + (mwk[m] == 0.f ? -100.f : 0.f);
        sv[i] = v; mx = fmaxf(mx, v);
      }
      red[row*4 + q4] = mx;
      __syncthreads();
      const float rm = fmaxf(fmaxf(red[row*4], red[row*4+1]),
                             fmaxf(red[row*4+2], red[row*4+3]));
      __syncthreads();
      float ssum = 0.f;
      #pragma unroll
      for (int i = 0; i < 16; ++i) { sv[i] = __expf(sv[i] - rm); ssum += sv[i]; }
      red[row*4 + q4] = ssum;
      __syncthreads();
      const float ir = 1.f / (red[row*4] + red[row*4+1] + red[row*4+2] + red[row*4+3]);
      #pragma unroll
      for (int i = 0; i < 16; ++i)
        ps16[row * NTOK + q4*16 + i] = (_Float16)(sv[i] * ir);
      __syncthreads();
    }

    // O_h = P V : 8 tiles (4 Mtiles x 2 Ntiles), K=64 -> 2 steps
    {
      const int mt = wv >> 1, nt = wv & 1;
      v8f acc = {};
      #pragma unroll
      for (int ks = 0; ks < 2; ++ks) {
        v16h a  = ldfrag(ps16, mt * 16, ks * 32, NTOK, lo, hi);
        v16h bf = ldfrag(vhT,  nt * 16, ks * 32, NTOK, lo, hi);
        acc = wmma16(a, bf, acc);
      }
      #pragma unroll
      for (int r = 0; r < 8; ++r)
        o16[(mt*16 + r + hi*8) * CDIM + h * HDIM + nt*16 + lo] = (_Float16)acc[r];
      __syncthreads();
    }
  }

  // ---------------- Output projection: out = O @ Wp^T + bp ----------------
  {
    const void* wpsrc = ws16 + 3 * (CDIM * CDIM);   // pre-converted f16 Wp
    #pragma unroll
    for (int j = 0; j < 18; ++j) {                  // 4608 b128 chunks / 256 thr
      unsigned c = (unsigned)(tid + 256 * j) * 16u;
      async_b128(OFF_WP + c, c, wpsrc);
    }
    wait_async_0();
    __syncthreads();
  }

  float* outg = out + (size_t)b * NTOK * CDIM;
  for (int job = wv; job < 48; job += 8) {          // 4 Mtiles x 12 Ntiles
    const int mt = job / 12, nt = job % 12;
    v8f acc = {};
    #pragma unroll
    for (int ks = 0; ks < 6; ++ks) {
      v16h a  = ldfrag(o16,  mt * 16, ks * 32, CDIM, lo, hi);
      v16h bf = ldfrag(wp_s, nt * 16, ks * 32, CDIM, lo, hi);
      acc = wmma16(a, bf, acc);
    }
    const float bias = bp[nt * 16 + lo];
    #pragma unroll
    for (int r = 0; r < 8; ++r)
      outg[(size_t)(mt*16 + r + hi*8) * CDIM + nt*16 + lo] = acc[r] + bias;
  }

  // mw_new: any-valid-token per window, tiled over tokens
  if (tid < NTOK) mwout[(size_t)b * NTOK + tid] = *mwsum;
}

extern "C" void kernel_launch(void* const* d_in, const int* in_sizes, int n_in,
                              void* d_out, int out_size, void* d_ws, size_t ws_size,
                              hipStream_t stream) {
  const float* x    = (const float*)d_in[0];
  const float* mask = (const float*)d_in[1];
  const float* mw   = (const float*)d_in[2];
  const float* Wq   = (const float*)d_in[3];
  const float* bq   = (const float*)d_in[4];
  const float* Wk   = (const float*)d_in[5];
  const float* bk   = (const float*)d_in[6];
  const float* Wv   = (const float*)d_in[7];
  const float* bv   = (const float*)d_in[8];
  const float* Wp   = (const float*)d_in[9];
  const float* bp   = (const float*)d_in[10];
  _Float16* ws16 = (_Float16*)d_ws;                 // 4 x 192 x 192 f16 = 288KB

  cvt_weights<<<dim3(144), dim3(256), 0, stream>>>(Wq, Wk, Wv, Wp, ws16);

  float* out   = (float*)d_out;
  float* mwout = out + (size_t)4096 * 64 * 192;
  win_attn_kernel<<<dim3(4096), dim3(256), 0, stream>>>(
      x, mask, mw, bq, bk, bv, bp, ws16, out, mwout);
}
